// Combined_Distance_15195594293783
// MI455X (gfx1250) — compile-verified, matrix operand order checked by reading, not run on hardware
//
#include <hip/hip_runtime.h>

// ---------------------------------------------------------------------------
// Problem constants (from reference)
// ---------------------------------------------------------------------------
#define B_    8
#define N1_   2048
#define N2_   1152
#define K_    1024
#define GROUP 9

#define MT_   128          // n-rows per workgroup
#define NT_   144          // m-cols per workgroup (= lcm(16,9) -> 9 WMMA tiles)
#define KC_   32           // k-chunk (one bf16 WMMA K-step)
#define NKC_  (K_ / KC_)   // 32 chunks
#define APAD  40           // LDS row pad (halfs) -> 80B stride, spreads banks
#define DPAD  148          // epilogue LDS row pad (floats)

#define NGROUPS_TOTAL (B_ * N1_ * (N2_ / GROUP))   // 2,097,152

typedef __bf16 v16bf __attribute__((ext_vector_type(16)));
typedef float  v8f   __attribute__((ext_vector_type(8)));

struct StageMem {
    unsigned short A[MT_][APAD];   // 128x32 bf16 tile (padded, rows 16B aligned)
    unsigned short Bm[NT_][APAD];  // 144x32 bf16 tile (padded)
};
union SMemU {
    StageMem s[2];                 // 43,520 B (double-buffered k-loop staging)
    float D[8][8][DPAD];           // 37,888 B (epilogue, per-wave 8x144 halves)
};

// ---- f32 -> packed bf16 (hardware cvt if available) -----------------------
#if __has_builtin(__builtin_amdgcn_cvt_pk_bf16_f32)
__device__ __forceinline__ unsigned pk_bf16(float a, float b) {
    auto r = __builtin_amdgcn_cvt_pk_bf16_f32(a, b);   // v_cvt_pk_bf16_f32
    return __builtin_bit_cast(unsigned, r);
}
#else
__device__ __forceinline__ unsigned pk_bf16(float a, float b) {
    unsigned ua = __float_as_uint(a), ub = __float_as_uint(b);
    ua = (ua + 0x7FFFu + ((ua >> 16) & 1u)) >> 16;
    ub = (ub + 0x7FFFu + ((ub >> 16) & 1u)) >> 16;
    return ua | (ub << 16);
}
#endif

// ---- 16B-vectorized LDS fragment load: 2x ds_load_b128 -> v16bf -----------
struct Frag32B { uint4 lo; uint4 hi; };
__device__ __forceinline__ v16bf load_frag(const unsigned short* rowPtr, int kb) {
    Frag32B t;
    t.lo = *(const uint4*)(rowPtr + kb);        // K group 0 (8 bf16, 16 B)
    t.hi = *(const uint4*)(rowPtr + kb + 16);   // K group 1 (8 bf16, 16 B)
    return __builtin_bit_cast(v16bf, t);
}

// ---- 16 f32 -> 16 bf16 -> 2x ds_store_b128 --------------------------------
__device__ __forceinline__ void store16(unsigned short* dst, const float4& f0,
                                        const float4& f1, const float4& f2,
                                        const float4& f3) {
    uint4 w0, w1;
    w0.x = pk_bf16(f0.x, f0.y); w0.y = pk_bf16(f0.z, f0.w);
    w0.z = pk_bf16(f1.x, f1.y); w0.w = pk_bf16(f1.z, f1.w);
    w1.x = pk_bf16(f2.x, f2.y); w1.y = pk_bf16(f2.z, f2.w);
    w1.z = pk_bf16(f3.x, f3.y); w1.w = pk_bf16(f3.z, f3.w);
    *(uint4*)dst       = w0;
    *((uint4*)dst + 1) = w1;
}

// ---------------------------------------------------------------------------
// Kernel 1: per-row squared norms (one wave32 per row)
// ---------------------------------------------------------------------------
__global__ __launch_bounds__(256)
void row_sqnorm(const float* __restrict__ x, float* __restrict__ out, int nrows) {
    const int row  = blockIdx.x * 8 + (threadIdx.x >> 5);
    const int lane = threadIdx.x & 31;
    if (row >= nrows) return;
    const float4* p = (const float4*)(x + (size_t)row * K_);
    float s = 0.0f;
    #pragma unroll
    for (int i = 0; i < K_ / 4 / 32; ++i) {            // 8 iters, 512B/wave each
        float4 v = p[lane + 32 * i];
        s = fmaf(v.x, v.x, s); s = fmaf(v.y, v.y, s);
        s = fmaf(v.z, v.z, s); s = fmaf(v.w, v.w, s);
    }
    #pragma unroll
    for (int m = 16; m >= 1; m >>= 1) s += __shfl_xor(s, m, 32);
    if (lane == 0) out[row] = s;
}

// ---------------------------------------------------------------------------
// Kernel 2: tiled bf16-WMMA cross term + fused norms + group-of-9 min + sum
// ---------------------------------------------------------------------------
__global__ __launch_bounds__(256)
void dist_min_kernel(const float* __restrict__ x1, const float* __restrict__ x2,
                     const float* __restrict__ sq1, const float* __restrict__ sq2,
                     float* __restrict__ partials) {
    __shared__ SMemU sm;
    __shared__ float sq1s[MT_];
    __shared__ float sq2s[NT_];
    __shared__ float wsum[8];

    const int tid   = threadIdx.x;
    const int wave  = tid >> 5;
    const int lane  = tid & 31;
    const int b     = blockIdx.z;
    const int nBase = blockIdx.y * MT_;
    const int mBase = blockIdx.x * NT_;

    if (tid < MT_) sq1s[tid] = sq1[b * N1_ + nBase + tid];
    if (tid < NT_) sq2s[tid] = sq2[b * N2_ + mBase + tid];

    // A-stage mapping: 2 threads per row (16 f32 each)
    const int rowA  = tid >> 1;
    const int halfA = tid & 1;
    const float* gA = x1 + ((size_t)(b * N1_ + nBase + rowA)) * K_ + halfA * 16;

    // B-stage mapping: 288 (row,half) pairs over 256 threads
    const int rB0 = tid >> 1,         hB0 = tid & 1;
    const int rB1 = 128 + (tid >> 1), hB1 = tid & 1;
    const bool hasB1 = (tid < 32);
    const float* gB0 = x2 + ((size_t)(b * N2_ + mBase + rB0)) * K_ + hB0 * 16;
    const float* gB1 = x2 + ((size_t)(b * N2_ + mBase + rB1)) * K_ + hB1 * 16;

    // Fragment addressing per ISA 7.12.2 (16-bit A/B):
    // lanes 0-15: K 0-7 then 16-23; lanes 16-31: K 8-15 then 24-31.
    const int fragRow = lane & 15;
    const int kb      = (lane >> 4) * 8;

    v8f acc[9];
    #pragma unroll
    for (int t = 0; t < 9; ++t)
        #pragma unroll
        for (int e = 0; e < 8; ++e) acc[t][e] = 0.0f;

    // ---- prologue: stage chunk 0 into buffer 0 ----
    {
        const float4* pA = (const float4*)gA;
        store16(&sm.s[0].A[rowA][halfA * 16], pA[0], pA[1], pA[2], pA[3]);
        const float4* pB0 = (const float4*)gB0;
        store16(&sm.s[0].Bm[rB0][hB0 * 16], pB0[0], pB0[1], pB0[2], pB0[3]);
        if (hasB1) {
            const float4* pB1 = (const float4*)gB1;
            store16(&sm.s[0].Bm[rB1][hB1 * 16], pB1[0], pB1[1], pB1[2], pB1[3]);
        }
    }

    for (int kc = 0; kc < NKC_; ++kc) {
        __syncthreads();                  // buf[cur] visible; buf[nxt] free
        const int cur = kc & 1;
        const int nxt = cur ^ 1;

        // ---- stage chunk kc+1 into the other buffer (overlaps WMMA) ----
        if (kc + 1 < NKC_) {
            const size_t ko = (size_t)(kc + 1) * KC_;
            const float4* pA = (const float4*)(gA + ko);
            float4 a0 = pA[0], a1 = pA[1], a2 = pA[2], a3 = pA[3];
            const float4* pB0 = (const float4*)(gB0 + ko);
            float4 b00 = pB0[0], b01 = pB0[1], b02 = pB0[2], b03 = pB0[3];
            if (kc + 2 < NKC_) {          // global_prefetch for chunk kc+2
                __builtin_prefetch(gA  + (size_t)(kc + 2) * KC_, 0, 0);
                __builtin_prefetch(gB0 + (size_t)(kc + 2) * KC_, 0, 0);
            }
            store16(&sm.s[nxt].A[rowA][halfA * 16], a0, a1, a2, a3);
            store16(&sm.s[nxt].Bm[rB0][hB0 * 16], b00, b01, b02, b03);
            if (hasB1) {
                const float4* pB1 = (const float4*)(gB1 + ko);
                store16(&sm.s[nxt].Bm[rB1][hB1 * 16], pB1[0], pB1[1], pB1[2], pB1[3]);
            }
        }

        // ---- compute chunk kc: 1 A fragment, 9 B fragments, 9 WMMAs ----
        v16bf afrag = load_frag(&sm.s[cur].A[(wave << 4) + fragRow][0], kb);
        v16bf bnext = load_frag(&sm.s[cur].Bm[fragRow][0], kb);   // ti = 0
        #pragma unroll
        for (int ti = 0; ti < 9; ++ti) {
            v16bf bcur = bnext;
            if (ti + 1 < 9)               // pipeline: issue loads before WMMA
                bnext = load_frag(&sm.s[cur].Bm[(ti + 1) * 16 + fragRow][0], kb);
            acc[ti] = __builtin_amdgcn_wmma_f32_16x16x32_bf16(
                false, afrag, false, bcur, (short)0, acc[ti], false, false);
        }
    }

    // ---- epilogue: d = sq1 + sq2 - 2*cross; min over groups of 9; sum ----
    __syncthreads();                      // staging LDS dead; reuse union as D
    float lsum = 0.0f;
    const int h   = lane >> 4;
    const int nlo = lane & 15;

    #pragma unroll
    for (int p = 0; p < 2; ++p) {         // two row-half passes
        #pragma unroll
        for (int ti = 0; ti < 9; ++ti) {
            #pragma unroll
            for (int j = 0; j < 4; ++j) {
                const int jj = p * 4 + j;            // C VGPR index
                const int rowStrip = jj + 8 * h;     // row within 16-row strip
                const int rrow = j + 4 * h;          // 0..7 in pass buffer
                const int col  = ti * 16 + nlo;      // 0..143
                float d = sq1s[(wave << 4) + rowStrip] + sq2s[col]
                          - 2.0f * acc[ti][jj];
                sm.D[wave][rrow][col] = d;
            }
        }
        __syncthreads();
        #pragma unroll
        for (int i = 0; i < 4; ++i) {                // 8 rows x 16 groups / 32 lanes
            const int idx = lane + 32 * i;
            const int rr = idx >> 4, g = idx & 15;
            const float* q = &sm.D[wave][rr][g * GROUP];
            float m = q[0];
            #pragma unroll
            for (int j = 1; j < GROUP; ++j) m = fminf(m, q[j]);
            lsum += m;
        }
        __syncthreads();
    }

    #pragma unroll
    for (int m = 16; m >= 1; m >>= 1) lsum += __shfl_xor(lsum, m, 32);
    if (lane == 0) wsum[wave] = lsum;
    __syncthreads();
    if (tid == 0) {
        float s = 0.0f;
        #pragma unroll
        for (int w = 0; w < 8; ++w) s += wsum[w];
        const int wg = (blockIdx.z * gridDim.y + blockIdx.y) * gridDim.x + blockIdx.x;
        partials[wg] = s;                 // deterministic: no atomics
    }
}

// ---------------------------------------------------------------------------
// Kernel 3: deterministic final reduction of 1024 workgroup partials
// ---------------------------------------------------------------------------
__global__ __launch_bounds__(256)
void final_reduce(const float* __restrict__ partials, float* __restrict__ out) {
    __shared__ float red[256];
    const int t = threadIdx.x;
    float s = partials[t] + partials[t + 256] + partials[t + 512] + partials[t + 768];
    red[t] = s;
    __syncthreads();
    #pragma unroll
    for (int off = 128; off > 0; off >>= 1) {
        if (t < off) red[t] += red[t + off];
        __syncthreads();
    }
    if (t == 0) out[0] = red[0] * (1.0f / (float)NGROUPS_TOTAL);
}

// ---------------------------------------------------------------------------
extern "C" void kernel_launch(void* const* d_in, const int* in_sizes, int n_in,
                              void* d_out, int out_size, void* d_ws, size_t ws_size,
                              hipStream_t stream) {
    const float* x1 = (const float*)d_in[0];   // [8, 2048, 1024] f32
    const float* x2 = (const float*)d_in[1];   // [8, 1152, 1024] f32

    // workspace layout: sq1 (64 KB) | sq2 (36 KB) | partials (4 KB)
    float* sq1      = (float*)d_ws;
    float* sq2      = (float*)((char*)d_ws + (size_t)B_ * N1_ * sizeof(float));
    float* partials = (float*)((char*)d_ws + (size_t)B_ * N1_ * sizeof(float)
                                           + (size_t)B_ * N2_ * sizeof(float));

    row_sqnorm<<<(B_ * N1_) / 8, 256, 0, stream>>>(x1, sq1, B_ * N1_);
    row_sqnorm<<<(B_ * N2_) / 8, 256, 0, stream>>>(x2, sq2, B_ * N2_);

    dim3 grid(N2_ / NT_, N1_ / MT_, B_);       // (8, 16, 8) = 1024 WGs
    dist_min_kernel<<<grid, 256, 0, stream>>>(x1, x2, sq1, sq2, partials);

    final_reduce<<<1, 256, 0, stream>>>(partials, (float*)d_out);
}
